// JambaSparseMoeBlock_867583393900
// MI455X (gfx1250) — matrix operand order repaired and checked by reading, not compiled, not run on hardware
//
#include <hip/hip_runtime.h>
#include <hip/hip_bf16.h>

// ---------------- problem constants ----------------
#define T_TOK 8192     // B*S tokens
#define H_DIM 1024
#define F_DIM 4096
#define E_NUM 8
#define MT    32       // token tile: two 16-row WMMA M-tiles (B fragments reused x2)
#define FC    512      // F chunk held in LDS per iteration
#define XS_STRIDE (H_DIM + 8)
#define HS_STRIDE (FC + 8)

// ---------------- WMMA types ----------------
typedef __bf16 bf16_t;
typedef bf16_t v16bf __attribute__((ext_vector_type(16)));
typedef float  v8f   __attribute__((ext_vector_type(8)));
typedef unsigned short ushort16v __attribute__((ext_vector_type(16)));
typedef unsigned short ushort8v  __attribute__((ext_vector_type(8)));

__device__ __forceinline__ unsigned short f2bf(float f) {
  bf16_t b = (bf16_t)f;                       // native f32->bf16 cvt (RNE)
  return __builtin_bit_cast(unsigned short, b);
}

__device__ __forceinline__ v8f wmma_bf16(v16bf a, v16bf b, v8f c) {
  return __builtin_amdgcn_wmma_f32_16x16x32_bf16(
      /*neg_a=*/false, a, /*neg_b=*/false, b,
      /*c_mod=*/(short)0, c, /*reuse_a=*/false, /*reuse_b=*/false);
}

// B fragment: per lane a contiguous 16-element (32B) run of one weight row.
__device__ __forceinline__ v16bf load_b16(const unsigned short* p) {
  ushort16v u = *(const ushort16v*)p;
  return __builtin_bit_cast(v16bf, u);
}

// A fragment from LDS: one contiguous 16-element run of row m (k = k0 + 2*klo + s).
__device__ __forceinline__ v16bf load_a_lds(const unsigned short* row, int k0, int klo) {
  ushort8v lo = *(const ushort8v*)(row + k0 + 2 * klo);
  ushort8v hi = *(const ushort8v*)(row + k0 + 2 * klo + 8);
  ushort16v u;
#pragma unroll
  for (int j = 0; j < 8; ++j) { u[j] = lo[j]; u[8 + j] = hi[j]; }
  return __builtin_bit_cast(v16bf, u);
}

// ---------------- kernel 1: zero per-expert counters ----------------
__global__ void moe_zero_kernel(int* __restrict__ counts) {
  if (threadIdx.x < E_NUM) counts[threadIdx.x] = 0;
}

// ---------------- kernel 2: router (one wave32 per token) ----------------
__global__ void moe_router_kernel(const float* __restrict__ x,
                                  const float* __restrict__ rw,
                                  unsigned short* __restrict__ Xb,
                                  int* __restrict__ counts,
                                  int* __restrict__ tok_idx,
                                  float* __restrict__ tok_w,
                                  unsigned int* __restrict__ codes) {
  const int gtid = blockIdx.x * blockDim.x + threadIdx.x;
  const int t    = gtid >> 5;
  const int lane = threadIdx.x & 31;
  if (t >= T_TOK) return;

  float xr[32];
#pragma unroll
  for (int i = 0; i < 32; ++i) {
    float v = x[(size_t)t * H_DIM + lane + 32 * i];
    xr[i] = v;
    Xb[(size_t)t * H_DIM + lane + 32 * i] = f2bf(v);   // bf16 activation copy
  }

  float logits[E_NUM];
#pragma unroll
  for (int e = 0; e < E_NUM; ++e) {
    float p = 0.0f;
#pragma unroll
    for (int i = 0; i < 32; ++i)
      p += xr[i] * rw[(size_t)e * H_DIM + lane + 32 * i];
#pragma unroll
    for (int m = 16; m >= 1; m >>= 1) p += __shfl_xor(p, m, 32);
    logits[e] = p;
  }

  if (lane == 0) {
    float mx = logits[0];
#pragma unroll
    for (int e = 1; e < E_NUM; ++e) mx = fmaxf(mx, logits[e]);
    float ex[E_NUM], s = 0.0f;
#pragma unroll
    for (int e = 0; e < E_NUM; ++e) { ex[e] = __expf(logits[e] - mx); s += ex[e]; }
    int i1 = 0;
#pragma unroll
    for (int e = 1; e < E_NUM; ++e) if (logits[e] > logits[i1]) i1 = e;
    int i2 = (i1 == 0) ? 1 : 0;
#pragma unroll
    for (int e = 0; e < E_NUM; ++e)
      if (e != i1 && logits[e] > logits[i2]) i2 = e;
    float inv = 1.0f / s;                    // Jamba: softmax probs, NOT renormalized over top-k
    int p1 = atomicAdd(&counts[i1], 1);
    tok_idx[i1 * T_TOK + p1] = t;  tok_w[i1 * T_TOK + p1] = ex[i1] * inv;
    int p2 = atomicAdd(&counts[i2], 1);
    tok_idx[i2 * T_TOK + p2] = t;  tok_w[i2 * T_TOK + p2] = ex[i2] * inv;
    codes[2 * t]     = ((unsigned)i1 << 16) | (unsigned)p1;  // where to gather from
    codes[2 * t + 1] = ((unsigned)i2 << 16) | (unsigned)p2;
  }
}

// ---------------- kernel 3: exclusive prefix of counts -> slot bases ----------------
__global__ void moe_prefix_kernel(const int* __restrict__ counts, int* __restrict__ base) {
  if (threadIdx.x == 0) {
    int s = 0;
#pragma unroll
    for (int e = 0; e < E_NUM; ++e) { base[e] = s; s += counts[e]; }
  }
}

// ---------------- kernel 4: weight fp32 -> bf16 ----------------
__global__ void moe_cvtw_kernel(const float* __restrict__ g, const float* __restrict__ u,
                                const float* __restrict__ d,
                                unsigned short* __restrict__ Gb, unsigned short* __restrict__ Ub,
                                unsigned short* __restrict__ Db, size_t nelem) {
  size_t i = (size_t)blockIdx.x * blockDim.x + threadIdx.x;
  size_t stride = (size_t)gridDim.x * blockDim.x;
  for (; i < nelem; i += stride) {
    Gb[i] = f2bf(g[i]);
    Ub[i] = f2bf(u[i]);
    Db[i] = f2bf(d[i]);
  }
}

// ---------------- kernel 5: fused expert FFN (grouped GEMM, bf16 WMMA) ----------------
__global__ void __launch_bounds__(256, 1)
moe_ffn_kernel(const unsigned short* __restrict__ Xb,
               const unsigned short* __restrict__ Gb,
               const unsigned short* __restrict__ Ub,
               const unsigned short* __restrict__ Db,
               const int* __restrict__ counts,
               const int* __restrict__ base,
               const int* __restrict__ tok_idx,
               const float* __restrict__ tok_w,
               float* __restrict__ pbuf) {
  const int e    = blockIdx.y;
  const int cnt  = counts[e];
  const int row0 = blockIdx.x * MT;
  if (row0 >= cnt) return;
  const int gb   = base[e];

  extern __shared__ __align__(32) char smem[];
  unsigned short* Xs = (unsigned short*)smem;          // [MT][XS_STRIDE] bf16 tokens
  unsigned short* Hs = Xs + MT * XS_STRIDE;            // [MT][HS_STRIDE] bf16 hid chunk
  int*   tks = (int*)(Hs + MT * HS_STRIDE);
  float* wks = (float*)(tks + MT);

  const int tid  = threadIdx.x;
  const int lane = tid & 31;
  const int wid  = tid >> 5;

  if (tid < MT) {
    int r = row0 + tid;
    int tv; float wv;
    if (r < cnt) { tv = tok_idx[e * T_TOK + r]; wv = tok_w[e * T_TOK + r]; }
    else         { tv = tok_idx[e * T_TOK + row0]; wv = 0.0f; }  // padded row (stores guarded)
    tks[tid] = tv;  wks[tid] = wv;
  }
  __syncthreads();

  // stage the MT token rows (bf16) into LDS, 16B vector copies
  for (int i = tid; i < MT * (H_DIM / 8); i += 256) {
    int r  = i >> 7;            // / (H_DIM/8 = 128)
    int c8 = (i & 127) * 8;
    *(ushort8v*)&Xs[r * XS_STRIDE + c8] = *(const ushort8v*)(Xb + (size_t)tks[r] * H_DIM + c8);
  }
  __syncthreads();

  const size_t wbase = (size_t)e * F_DIM * H_DIM;  // gate/up: [F,H]
  const size_t dbase = (size_t)e * H_DIM * F_DIM;  // down:    [H,F]

  // ISA fragment lane decomposition
  const int n   = lane & 15;          // B/D column
  const int hiw = lane >> 4;          // half-wave select
  const int klo = hiw * 8;            // A: K sub-offset
  const int kb  = hiw * 16;           // B: K sub-offset
  const int mb  = hiw * 8;            // C/D: row base

  v8f acc[16];                        // out tile [32, wid*128 .. +128) fp32
#pragma unroll
  for (int i = 0; i < 16; ++i) acc[i] = (v8f){};

  for (int fc = 0; fc < F_DIM / FC; ++fc) {
    const int f0 = fc * FC;

    // ---- phase 1: gate & up for columns [wid*64, wid*64+64), silu*up -> Hs ----
#pragma unroll
    for (int nt = 0; nt < 4; ++nt) {
      const int fcol = f0 + wid * 64 + nt * 16 + n;     // weight row (B column) for this lane
      const unsigned short* gp = Gb + wbase + (size_t)fcol * H_DIM + kb;
      const unsigned short* up = Ub + wbase + (size_t)fcol * H_DIM + kb;
      __builtin_prefetch(gp, 0, 1);
      __builtin_prefetch(up, 0, 1);
      v8f g0 = (v8f){}, u0 = (v8f){};
      v8f g1 = (v8f){}, u1 = (v8f){};
#pragma unroll 4
      for (int k0 = 0; k0 < H_DIM; k0 += 32) {
        v16bf a0 = load_a_lds(Xs + n * XS_STRIDE,        k0, klo);
        v16bf a1 = load_a_lds(Xs + (16 + n) * XS_STRIDE, k0, klo);
        v16bf bg = load_b16(gp + k0);
        v16bf bu = load_b16(up + k0);
        g0 = wmma_bf16(a0, bg, g0);
        g1 = wmma_bf16(a1, bg, g1);     // B fragment reused across both M-tiles
        u0 = wmma_bf16(a0, bu, u0);
        u1 = wmma_bf16(a1, bu, u1);
      }
      const int col = wid * 64 + nt * 16 + n;
#pragma unroll
      for (int r = 0; r < 8; ++r) {
        float gv0 = g0[r];
        float hv0 = gv0 * __builtin_amdgcn_rcpf(1.0f + __expf(-gv0)) * u0[r];  // silu(g)*u
        Hs[(mb + r) * HS_STRIDE + col] = f2bf(hv0);
        float gv1 = g1[r];
        float hv1 = gv1 * __builtin_amdgcn_rcpf(1.0f + __expf(-gv1)) * u1[r];
        Hs[(16 + mb + r) * HS_STRIDE + col] = f2bf(hv1);
      }
    }
    __syncthreads();

    // ---- phase 2: down-projection partial, K = this F chunk ----
#pragma unroll
    for (int nt = 0; nt < 8; ++nt) {
      const int hcol = wid * 128 + nt * 16 + n;         // down_w row (B column)
      const unsigned short* dp = Db + dbase + (size_t)hcol * F_DIM + f0 + kb;
      __builtin_prefetch(dp, 0, 1);
      v8f c0 = acc[nt];
      v8f c1 = acc[8 + nt];
#pragma unroll 4
      for (int k0 = 0; k0 < FC; k0 += 32) {
        v16bf a0 = load_a_lds(Hs + n * HS_STRIDE,        k0, klo);
        v16bf a1 = load_a_lds(Hs + (16 + n) * HS_STRIDE, k0, klo);
        v16bf bv = load_b16(dp + k0);
        c0 = wmma_bf16(a0, bv, c0);
        c1 = wmma_bf16(a1, bv, c1);
      }
      acc[nt] = c0;
      acc[8 + nt] = c1;
    }
    __syncthreads();
  }

  // ---- epilogue: scale by routing weight, plain coalesced stores to slot buffer ----
#pragma unroll
  for (int nt = 0; nt < 8; ++nt) {
    const int h = wid * 128 + nt * 16 + n;
#pragma unroll
    for (int r = 0; r < 8; ++r) {
      const int row = row0 + mb + r;
      if (row < cnt)
        pbuf[(size_t)(gb + row) * H_DIM + h] = wks[mb + r] * acc[nt][r];
      const int row1 = row0 + 16 + mb + r;
      if (row1 < cnt)
        pbuf[(size_t)(gb + row1) * H_DIM + h] = wks[16 + mb + r] * acc[8 + nt][r];
    }
  }
}

// ---------------- kernel 6: gather-combine (out = pbuf[slotA] + pbuf[slotB]) ----------------
__global__ void moe_combine_kernel(const float* __restrict__ pbuf,
                                   const unsigned int* __restrict__ codes,
                                   const int* __restrict__ base,
                                   float* __restrict__ out) {
  const int idx = blockIdx.x * blockDim.x + threadIdx.x;   // over T*H/4
  const int t   = idx >> 8;                                // H/4 = 256 per token
  const int h4  = (idx & 255) * 4;
  const unsigned int c0 = codes[2 * t];
  const unsigned int c1 = codes[2 * t + 1];
  const size_t s0 = (size_t)(base[c0 >> 16] + (int)(c0 & 0xFFFFu)) * H_DIM + h4;
  const size_t s1 = (size_t)(base[c1 >> 16] + (int)(c1 & 0xFFFFu)) * H_DIM + h4;
  const float4 a = *(const float4*)(pbuf + s0);
  const float4 b = *(const float4*)(pbuf + s1);
  float4 o;
  o.x = a.x + b.x;  o.y = a.y + b.y;  o.z = a.z + b.z;  o.w = a.w + b.w;
  *(float4*)(out + (size_t)t * H_DIM + h4) = o;
}

// ---------------- launcher ----------------
extern "C" void kernel_launch(void* const* d_in, const int* in_sizes, int n_in,
                              void* d_out, int out_size, void* d_ws, size_t ws_size,
                              hipStream_t stream) {
  const float* x  = (const float*)d_in[0];   // [B,S,H]
  const float* rw = (const float*)d_in[1];   // [E,H]
  const float* gw = (const float*)d_in[2];   // [E,F,H]
  const float* uw = (const float*)d_in[3];   // [E,F,H]
  const float* dw = (const float*)d_in[4];   // [E,H,F]
  float* out = (float*)d_out;                // [B,S,H] fp32

  char* ws = (char*)d_ws;
  size_t off = 0;
  auto take = [&](size_t bytes) -> char* {
    char* p = ws + off;
    off = (off + bytes + 255) & ~(size_t)255;
    return p;
  };
  unsigned short* Xb = (unsigned short*)take((size_t)T_TOK * H_DIM * 2);
  unsigned short* Gb = (unsigned short*)take((size_t)E_NUM * F_DIM * H_DIM * 2);
  unsigned short* Ub = (unsigned short*)take((size_t)E_NUM * F_DIM * H_DIM * 2);
  unsigned short* Db = (unsigned short*)take((size_t)E_NUM * H_DIM * F_DIM * 2);
  int*            cn = (int*)take(E_NUM * sizeof(int));
  int*            bs = (int*)take(E_NUM * sizeof(int));
  int*            ti = (int*)take((size_t)E_NUM * T_TOK * sizeof(int));
  float*          tw = (float*)take((size_t)E_NUM * T_TOK * sizeof(float));
  unsigned int*   cd = (unsigned int*)take((size_t)2 * T_TOK * sizeof(unsigned int));
  float*          pb = (float*)take((size_t)2 * T_TOK * H_DIM * sizeof(float)); // 67 MB slots

  // 1) zero per-expert counters
  moe_zero_kernel<<<1, 32, 0, stream>>>(cn);

  // 2) router + top-2 compaction + gather codes + x -> bf16
  moe_router_kernel<<<(T_TOK * 32) / 256, 256, 0, stream>>>(x, rw, Xb, cn, ti, tw, cd);

  // 3) slot bases (exclusive prefix over 8 experts)
  moe_prefix_kernel<<<1, 32, 0, stream>>>(cn, bs);

  // 4) weights fp32 -> bf16
  moe_cvtw_kernel<<<8192, 256, 0, stream>>>(gw, uw, dw, Gb, Ub, Db,
                                            (size_t)E_NUM * F_DIM * H_DIM);

  // 5) fused grouped-GEMM FFN, WMMA bf16, dynamic LDS (~97 KB of the 320 KB WGP LDS)
  size_t shmem = (size_t)MT * XS_STRIDE * 2 + (size_t)MT * HS_STRIDE * 2
               + MT * sizeof(int) + MT * sizeof(float);
  dim3 grid(T_TOK / MT, E_NUM);
  moe_ffn_kernel<<<grid, 256, shmem, stream>>>(Xb, Gb, Ub, Db, cn, bs, ti, tw, pb);

  // 6) streaming top-2 combine (fully overwrites d_out; no atomics, no pre-zero)
  moe_combine_kernel<<<(T_TOK * (H_DIM / 4)) / 256, 256, 0, stream>>>(pb, cd, bs, out);
}
